// Encoder_45784351375645
// MI455X (gfx1250) — matrix operand order repaired and checked by reading, not compiled, not run on hardware
//
#include <hip/hip_runtime.h>
#include <hip/hip_bf16.h>

typedef __attribute__((ext_vector_type(16))) __bf16 v16bf;
typedef __attribute__((ext_vector_type(8)))  __bf16 v8bf;
typedef __attribute__((ext_vector_type(8)))  float  v8f;

// ---------------- fast device math ----------------
__device__ __forceinline__ float fast_tanh(float x) {
#if __has_builtin(__builtin_amdgcn_tanhf)
  return __builtin_amdgcn_tanhf(x);            // v_tanh_f32 (gfx1250 trans op)
#else
  float e = __expf(2.0f * x);
  return 1.0f - 2.0f / (e + 1.0f);
#endif
}
__device__ __forceinline__ float fast_sigmoid(float x) {
  return 1.0f / (1.0f + __expf(-x));           // v_exp_f32 + rcp
}

// ---------------- WMMA fragment loaders (CDNA5 16-bit layouts) ----------------
// A (16x32 bf16): lane holds row m=lane&15; elements 0..7 at k = kk+8*hi,
// elements 8..15 at k = kk+16+8*hi (hi = lane>>4).
__device__ __forceinline__ v16bf loadA_bf16(const __bf16* row, int kk, int hi) {
  v8bf lo = *(const v8bf*)(row + kk + 8 * hi);
  v8bf hv = *(const v8bf*)(row + kk + 16 + 8 * hi);
  return __builtin_shufflevector(lo, hv, 0,1,2,3,4,5,6,7,8,9,10,11,12,13,14,15);
}
// B (32x16 bf16): lane holds column n=lane&15; elements 0..7 at k = kk+8*hi,
// elements 8..15 at k = kk+16+8*hi (same order as A).
__device__ __forceinline__ v16bf loadB_bf16(const __bf16* row, int kk, int hi) {
  v8bf lo = *(const v8bf*)(row + kk + 8 * hi);
  v8bf hv = *(const v8bf*)(row + kk + 16 + 8 * hi);
  return __builtin_shufflevector(lo, hv, 0,1,2,3,4,5,6,7,8,9,10,11,12,13,14,15);
}

// ---------------- kernel 0: weight conversion / bias fusion ----------------
__global__ void prep_weights(const float* __restrict__ w_ih, const float* __restrict__ w_hh,
                             const float* __restrict__ w_w,  const float* __restrict__ w_u,
                             const float* __restrict__ b_ih, const float* __restrict__ b_hh,
                             __bf16* __restrict__ wih_b, __bf16* __restrict__ whh_b,
                             __bf16* __restrict__ ww_b,  __bf16* __restrict__ wu_b,
                             float* __restrict__ bias_g) {
  int tid = blockIdx.x * blockDim.x + threadIdx.x;   // 65536 threads
  if (tid < 256 * 256) wih_b[tid] = (__bf16)w_ih[tid];
  if (tid < 256 * 64)  whh_b[tid] = (__bf16)w_hh[tid];
  if (tid < 64 * 128)  ww_b[tid]  = (__bf16)w_w[tid];
  if (tid < 64 * 64)   wu_b[tid]  = (__bf16)w_u[tid];
  if (tid < 256)       bias_g[tid] = b_ih[tid] + b_hh[tid];
}

// ---------------- kernel 1: u_out = x @ w_u^T + b_u   (M=B*V=131072, N=64, K=64) ----
__global__ __launch_bounds__(256) void uout_gemm(const float* __restrict__ x,
                                                 const __bf16* __restrict__ wu_b,
                                                 const float* __restrict__ b_u,
                                                 float* __restrict__ u_out) {
  const int wave   = threadIdx.x >> 5;
  const int lane   = threadIdx.x & 31;
  const int lanelo = lane & 15;
  const int hi     = lane >> 4;
  const int mt = blockIdx.x * 2 + (wave >> 2);       // 16-row M tile
  const int nt = wave & 3;                           // 16-col N tile (N=64)
  const float*  xr = x    + (size_t)(mt * 16 + lanelo) * 64;
  const __bf16* wr = wu_b + (nt * 16 + lanelo) * 64;
  v8f acc = {};
  #pragma unroll
  for (int kk = 0; kk < 64; kk += 32) {
    const int k0 = kk + 8 * hi;
    v16bf a;
    #pragma unroll
    for (int e = 0; e < 8; ++e) {
      a[e]     = (__bf16)xr[k0 + e];
      a[e + 8] = (__bf16)xr[k0 + 16 + e];
    }
    v16bf b = loadB_bf16(wr, kk, hi);
    acc = __builtin_amdgcn_wmma_f32_16x16x32_bf16(false, a, false, b, (short)0, acc, false, false);
  }
  const int col = nt * 16 + lanelo;
  const float bias = b_u[col];
  #pragma unroll
  for (int r = 0; r < 8; ++r) {
    int row = mt * 16 + 8 * hi + r;                  // C layout: M = r + 8*hi, N = lane&15
    u_out[(size_t)row * 64 + col] = acc[r] + bias;
  }
}

// ---------------- kernel 2: persistent 64-step attention-LSTM scan ----------------
// One block owns 16 batch rows for all 64 time steps. 32 waves (1024 threads).
__global__ __launch_bounds__(1024) void scan_kernel(
    const float* __restrict__ x,        // (512,256,64)
    const float* __restrict__ u_out,    // (512*256,64) f32 (L2-resident)
    const __bf16* __restrict__ wih_b,   // (256,256)  N x K
    const __bf16* __restrict__ whh_b,   // (256,64)   N x K
    const __bf16* __restrict__ ww_b,    // (64,128)   N x K
    const float* __restrict__ bias_g,   // (256) b_ih+b_hh
    const float* __restrict__ w_v,      // (64)
    const float* __restrict__ b_w,      // (64)
    float* __restrict__ out_xt,         // (64,512,256)
    float* __restrict__ out_h)          // (64,512,64)
{
  __shared__ float  c_lds[16][64];
  __shared__ __bf16 hc_bf[16][128];     // cols 0..63 = h, 64..127 = c (bf16, WMMA A)
  __shared__ float  wout[16][64];
  __shared__ float  score[16][256];     // scores then alphas
  __shared__ __bf16 xt_bf[16][256];
  __shared__ float  gates[16][256];
  __shared__ float  wv_l[64], bw_l[64], bg_l[256];

  const int tid    = threadIdx.x;
  const int wave   = tid >> 5;
  const int lane   = tid & 31;
  const int lanelo = lane & 15;
  const int hi     = lane >> 4;
  const int b0     = blockIdx.x * 16;

  for (int i = tid; i < 16 * 64;  i += 1024) c_lds[i >> 6][i & 63]  = 0.f;
  for (int i = tid; i < 16 * 128; i += 1024) hc_bf[i >> 7][i & 127] = (__bf16)0.f;
  if (tid < 64)  { wv_l[tid] = w_v[tid]; bw_l[tid] = b_w[tid]; }
  if (tid < 256) bg_l[tid] = bias_g[tid];
  __syncthreads();

  for (int t = 0; t < 64; ++t) {
    // stage 1: wout = [h|c](16x128) @ w_w^T + b_w   (waves 0..3, WMMA, K=128)
    if (wave < 4) {
      const __bf16* arow = &hc_bf[lanelo][0];
      const __bf16* brow = ww_b + (wave * 16 + lanelo) * 128;
      v8f acc = {};
      #pragma unroll
      for (int kk = 0; kk < 128; kk += 32) {
        v16bf a = loadA_bf16(arow, kk, hi);
        v16bf b = loadB_bf16(brow, kk, hi);
        acc = __builtin_amdgcn_wmma_f32_16x16x32_bf16(false, a, false, b, (short)0, acc, false, false);
      }
      const int col = wave * 16 + lanelo;
      #pragma unroll
      for (int r = 0; r < 8; ++r) wout[8 * hi + r][col] = acc[r] + bw_l[col];
    }
    __syncthreads();

    // stage 2: score[b][v] = sum_w tanh(wout[b][w] + u_out[b][v][w]) * w_v[w]
    // float4 global loads (b128), float4 LDS reads, prefetch of next row.
    #pragma unroll
    for (int j = 0; j < 4; ++j) {
      int p = tid + 1024 * j;                    // 4096 (b,v) pairs
      int b = p >> 8, v = p & 255;
      const float4* ur4 = (const float4*)(u_out + ((size_t)(b0 + b) * 256 + v) * 64);
      if (j < 3) {                               // prefetch next pair's u_out row
        int p2 = p + 1024;
        __builtin_prefetch(u_out + ((size_t)(b0 + (p2 >> 8)) * 256 + (p2 & 255)) * 64, 0, 3);
      }
      const float4* wo4 = (const float4*)(&wout[b][0]);
      const float4* wv4 = (const float4*)(&wv_l[0]);
      float s = 0.f;
      #pragma unroll 4
      for (int w4 = 0; w4 < 16; ++w4) {
        float4 u  = ur4[w4];
        float4 wo = wo4[w4];
        float4 wv = wv4[w4];
        s += fast_tanh(wo.x + u.x) * wv.x;
        s += fast_tanh(wo.y + u.y) * wv.y;
        s += fast_tanh(wo.z + u.z) * wv.z;
        s += fast_tanh(wo.w + u.w) * wv.w;
      }
      score[b][v] = s;
    }
    __syncthreads();

    // stage 3: softmax over v=256 (one wave per batch row)
    if (wave < 16) {
      float e[8];
      float m = -3.4e38f;
      #pragma unroll
      for (int j = 0; j < 8; ++j) { e[j] = score[wave][lane + 32 * j]; m = fmaxf(m, e[j]); }
      #pragma unroll
      for (int off = 16; off > 0; off >>= 1) m = fmaxf(m, __shfl_xor(m, off, 32));
      float sum = 0.f;
      #pragma unroll
      for (int j = 0; j < 8; ++j) { e[j] = __expf(e[j] - m); sum += e[j]; }
      #pragma unroll
      for (int off = 16; off > 0; off >>= 1) sum += __shfl_xor(sum, off, 32);
      float inv = 1.0f / sum;
      #pragma unroll
      for (int j = 0; j < 8; ++j) score[wave][lane + 32 * j] = e[j] * inv;
    }
    __syncthreads();

    // stage 4: x_tilde = alpha * x_t ; write output + bf16 copy for WMMA
    #pragma unroll
    for (int j = 0; j < 4; ++j) {
      int p = tid + 1024 * j;
      int b = p >> 8, v = p & 255;
      float xt = score[b][v] * x[((size_t)(b0 + b) * 256 + v) * 64 + t];
      out_xt[((size_t)t * 512 + b0 + b) * 256 + v] = xt;
      xt_bf[b][v] = (__bf16)xt;
    }
    __syncthreads();

    // stage 5: gates = x_tilde @ w_ih^T + h @ w_hh^T + bias   (waves 0..15, WMMA)
    if (wave < 16) {
      const __bf16* arow = &xt_bf[lanelo][0];
      const __bf16* hrow = &hc_bf[lanelo][0];         // h = cols 0..63
      const __bf16* brow1 = wih_b + (wave * 16 + lanelo) * 256;
      const __bf16* brow2 = whh_b + (wave * 16 + lanelo) * 64;
      v8f acc = {};
      #pragma unroll
      for (int kk = 0; kk < 256; kk += 32) {
        v16bf a = loadA_bf16(arow, kk, hi);
        v16bf b = loadB_bf16(brow1, kk, hi);
        acc = __builtin_amdgcn_wmma_f32_16x16x32_bf16(false, a, false, b, (short)0, acc, false, false);
      }
      #pragma unroll
      for (int kk = 0; kk < 64; kk += 32) {
        v16bf a = loadA_bf16(hrow, kk, hi);
        v16bf b = loadB_bf16(brow2, kk, hi);
        acc = __builtin_amdgcn_wmma_f32_16x16x32_bf16(false, a, false, b, (short)0, acc, false, false);
      }
      const int col = wave * 16 + lanelo;
      const float bias = bg_l[col];
      #pragma unroll
      for (int r = 0; r < 8; ++r) gates[8 * hi + r][col] = acc[r] + bias;
    }
    __syncthreads();

    // stage 6: LSTM pointwise update (1024 threads == 16*64 cells)
    {
      int b = tid >> 6, j = tid & 63;
      float ig = fast_sigmoid(gates[b][j]);
      float fg = fast_sigmoid(gates[b][64 + j]);
      float gg = fast_tanh(gates[b][128 + j]);
      float og = fast_sigmoid(gates[b][192 + j]);
      float cn = fg * c_lds[b][j] + ig * gg;
      float hn = og * fast_tanh(cn);
      c_lds[b][j]      = cn;
      hc_bf[b][j]      = (__bf16)hn;
      hc_bf[b][64 + j] = (__bf16)cn;
      out_h[((size_t)t * 512 + b0 + b) * 64 + j] = hn;
    }
    __syncthreads();
  }
}

// ---------------- launch ----------------
extern "C" void kernel_launch(void* const* d_in, const int* in_sizes, int n_in,
                              void* d_out, int out_size, void* d_ws, size_t ws_size,
                              hipStream_t stream) {
  const float* x    = (const float*)d_in[0];
  const float* w_ih = (const float*)d_in[1];
  const float* w_hh = (const float*)d_in[2];
  const float* b_ih = (const float*)d_in[3];
  const float* b_hh = (const float*)d_in[4];
  const float* w_v  = (const float*)d_in[5];
  const float* w_w  = (const float*)d_in[6];
  const float* b_w  = (const float*)d_in[7];
  const float* w_u  = (const float*)d_in[8];
  const float* b_u  = (const float*)d_in[9];

  char* ws = (char*)d_ws;
  __bf16* wih_b  = (__bf16*)(ws + 0);        // 256*256*2 = 131072
  __bf16* whh_b  = (__bf16*)(ws + 131072);   // 256*64*2  =  32768
  __bf16* ww_b   = (__bf16*)(ws + 163840);   // 64*128*2  =  16384
  __bf16* wu_b   = (__bf16*)(ws + 180224);   // 64*64*2   =   8192
  float*  bias_g = (float*) (ws + 188416);   // 256*4     =   1024
  float*  u_out  = (float*) (ws + 189440);   // 512*256*64*4 = 33.5 MB

  float* out_xt = (float*)d_out;                       // (64,512,256)
  float* out_h  = out_xt + (size_t)64 * 512 * 256;     // (64,512,64)

  prep_weights<<<256, 256, 0, stream>>>(w_ih, w_hh, w_w, w_u, b_ih, b_hh,
                                        wih_b, whh_b, ww_b, wu_b, bias_g);
  uout_gemm<<<4096, 256, 0, stream>>>(x, wu_b, b_u, u_out);
  scan_kernel<<<32, 1024, 0, stream>>>(x, u_out, wih_b, whh_b, ww_b, bias_g,
                                       w_v, b_w, out_xt, out_h);
}
